// GVNet_22522808500695
// MI455X (gfx1250) — compile-verified, matrix operand order checked by reading, not run on hardware
//
#include <hip/hip_runtime.h>
#include <hip/hip_bf16.h>

// ---------------------------------------------------------------------------
// GVNet forward for MI455X (gfx1250, wave32).
// Convs = implicit-GEMM on v_wmma_f32_16x16x32_f16, NHWC f16 activations,
// pre-packed f16 weights in (ky,kx,ci) K-order so im2col K-runs are
// contiguous. LDS staging is double-buffered and pipelined through the
// gfx1250 async global->LDS engine (global_load_async_to_lds_b128 +
// s_wait_asynccnt): chunk i+1 loads overlap chunk i WMMAs. GEMM is
// [pixels x channels] so NHWC stores are coalesced and BN folds to
// per-lane scalars.
//
// Assumed d_in flattening (depth-first, dict insertion order of setup_inputs):
//   per block bi (base = bi*20):
//     +0 w1, +1..4 bn1{g,b,m,v}, +5 w2, +6..9 bn2{g,b,m,v},
//     +10 w3, +11..14 bn3{g,b,m,v}, +15 wsc, +16..19 bnsc{g,b,m,v}
//   kp (base 80): 80 w1, 81..84 bn{g,b,m,v}, 85 w2, 86 b2
//   87 sx [25,3,84,84], 88 qx [75,3,84,84], 89 sy [25] int, 90 qy [75] int
// ---------------------------------------------------------------------------

typedef __attribute__((ext_vector_type(16))) _Float16 v16h;
typedef __attribute__((ext_vector_type(8)))  float    v8f;

#define GV_TAU 0.15f
#define F_LEAKY 1
#define F_BN    2
#define F_RES   4
#define F_BIAS  8

// ---- gfx1250 async global->LDS copy helpers (ASYNCcnt-tracked) ------------
__device__ __forceinline__ void gv_async_b128(void* lds_dst, const void* gsrc)
{
    // LDS aperture flat address truncates to the LDS byte offset (ISA 10.2).
    unsigned l = (unsigned)(unsigned long long)lds_dst;
    asm volatile("global_load_async_to_lds_b128 %0, %1, off"
                 :: "v"(l), "v"((unsigned long long)gsrc)
                 : "memory");
}
__device__ __forceinline__ void gv_async_wait0()
{
    asm volatile("s_wait_asynccnt 0x0" ::: "memory");
}

// ---------------------------------------------------------------------------
// Pack W f32 [Cout][Cin][ky][kx] -> f16 [Cout][(ky*ksz+kx)*Cin + ci]
__global__ void gv_pack_w(const float* __restrict__ W, _Float16* __restrict__ out,
                          int Cout, int Cin, int ksz)
{
    const int K = Cin * ksz * ksz;
    const long i = (long)blockIdx.x * blockDim.x + threadIdx.x;
    if (i >= (long)Cout * K) return;
    const int kp = (int)(i % K);
    const int co = (int)(i / K);
    const int ci = kp % Cin;
    const int rr = kp / Cin;             // ky*ksz + kx
    const int ky = rr / ksz, kx = rr - ky * ksz;
    out[i] = (_Float16)W[(((long)co * Cin + ci) * ksz + ky) * ksz + kx];
}

// ---------------------------------------------------------------------------
// Concat sx|qx and convert NCHW f32 -> NHWC f16  ([100,84,84,3])
__global__ void gv_to_nhwc(const float* __restrict__ sx, const float* __restrict__ qx,
                           _Float16* __restrict__ out)
{
    const long total = (long)100 * 84 * 84 * 3;
    const long i = (long)blockIdx.x * blockDim.x + threadIdx.x;
    if (i >= total) return;
    const int c  = (int)(i % 3);
    long r = i / 3;
    const int hw = (int)(r % 7056);
    const int b  = (int)(r / 7056);
    const float v = (b < 25) ? sx[((long)b * 3 + c) * 7056 + hw]
                             : qx[((long)(b - 25) * 3 + c) * 7056 + hw];
    out[i] = (_Float16)v;
}

// ---------------------------------------------------------------------------
// Implicit-GEMM conv, NHWC f16.  D[pixels, Cout] = im2col[pixels, K] x Wp^T.
// 256 threads = 8 waves; tile 128 pixels x 64 channels; BK=64 (2 WMMA steps).
// Wave w: pixel half mh=w>>2 (64 pixels = 4 accs), channel quarter nq=w&3.
// Double-buffered async staging: chunk i+1 DMA overlaps chunk i WMMAs.
// ---------------------------------------------------------------------------
__global__ __launch_bounds__(256)
void gv_conv_wmma(const _Float16* __restrict__ X,   // NHWC [B,H,W,Cin]
                  const _Float16* __restrict__ Wp,  // packed [Cout,K]
                  _Float16* __restrict__ Y,         // NHWC [B,H,W,Cout]
                  const float* __restrict__ bng, const float* __restrict__ bnb,
                  const float* __restrict__ bnm, const float* __restrict__ bnv,
                  const _Float16* __restrict__ res, const float* __restrict__ bias,
                  int B, int Cin, int H, int Wd, int Cout, int ksz,
                  int cinLog2 /* -1 if Cin not pow2 */, int flags)
{
    __shared__ _Float16 As[2][128][72];  // [buf][pixel][K]; row stride 144B=9*16B
    __shared__ _Float16 Bs[2][64][72];   // [buf][chan][K]

    const int tid  = threadIdx.x;
    const int lane = tid & 31;
    const int wave = tid >> 5;
    const int HW   = H * Wd;
    const int Npix = B * HW;
    const int K    = Cin * ksz * ksz;
    const int pad  = (ksz == 3) ? 1 : 0;
    const int m0   = blockIdx.x * 128;   // pixel tile
    const int n0   = blockIdx.y * 64;    // channel tile

    // ---- hoisted per-thread staging coordinates ----
    const int arow = tid >> 1;                 // As row (pixel within tile)
    const int akb  = (tid & 1) * 32;           // As K sub-run
    const int ap   = m0 + arow;
    const bool avalid = (ap < Npix);
    int ab = 0, ay = 0, ax = 0;
    if (avalid) {
        ab = ap / HW; const int hw = ap - ab * HW;
        ay = hw / Wd; ax = hw - ay * Wd;
    }
    const int brow = tid >> 2;                 // Bs row (channel within tile)
    const int bkb  = (tid & 3) * 16;
    const int bgm  = n0 + brow;
    const bool bvalid = (bgm < Cout);

    const bool fast = (cinLog2 >= 0);          // pow2 Cin: K multiple of 64
    const int nchunks = (K + 63) / 64;

    // ---- staging: issue async DMA (fast path) / scalar fill (edges) ----
    auto stage = [&](int k0, int buf) {
        {   // A tile (im2col, NHWC): 32 contiguous halves per thread
            _Float16* dst = &As[buf][arow][akb];
            const int gk0 = k0 + akb;
            if (fast) {
                if (avalid) {
                    const int rr  = gk0 >> cinLog2;
                    const int ci0 = gk0 & (Cin - 1);
                    const int ky  = (ksz == 3) ? (rr / 3) : 0;
                    const int kx  = (ksz == 3) ? (rr - ky * 3) : 0;
                    int yy = ay + ky - pad, xx = ax + kx - pad;
                    yy = yy < 0 ? -yy : (yy >= H  ? 2 * H  - 2 - yy : yy);
                    xx = xx < 0 ? -xx : (xx >= Wd ? 2 * Wd - 2 - xx : xx);
                    const _Float16* src =
                        X + ((long)(ab * HW) + yy * Wd + xx) * Cin + ci0;
                    gv_async_b128(dst,      src);       // 4 x 16B = 32 halves
                    gv_async_b128(dst + 8,  src + 8);
                    gv_async_b128(dst + 16, src + 16);
                    gv_async_b128(dst + 24, src + 24);
                } else {
                    #pragma unroll
                    for (int j = 0; j < 32; ++j) dst[j] = (_Float16)0.f;
                }
            } else {                          // generic (Cin==3, single chunk)
                for (int j = 0; j < 32; ++j) {
                    const int gk = gk0 + j;
                    float v = 0.f;
                    if (avalid && gk < K) {
                        const int rr = gk / Cin;
                        const int ci = gk - rr * Cin;
                        const int ky = rr / ksz, kx = rr - ky * ksz;
                        int yy = ay + ky - pad, xx = ax + kx - pad;
                        yy = yy < 0 ? -yy : (yy >= H  ? 2 * H  - 2 - yy : yy);
                        xx = xx < 0 ? -xx : (xx >= Wd ? 2 * Wd - 2 - xx : xx);
                        v = (float)X[((long)(ab * HW) + yy * Wd + xx) * Cin + ci];
                    }
                    dst[j] = (_Float16)v;
                }
            }
        }
        {   // B tile (packed weights): 16 contiguous halves per thread
            _Float16* dst = &Bs[buf][brow][bkb];
            if (fast && bvalid) {
                const _Float16* src = Wp + (long)bgm * K + k0 + bkb;
                gv_async_b128(dst,     src);            // 2 x 16B = 16 halves
                gv_async_b128(dst + 8, src + 8);
            } else if (bvalid) {
                const _Float16* src = Wp + (long)bgm * K + k0 + bkb;
                #pragma unroll
                for (int j = 0; j < 16; ++j) {
                    const int gk = k0 + bkb + j;
                    dst[j] = (gk < K) ? src[j] : (_Float16)0.f;
                }
            } else {
                #pragma unroll
                for (int j = 0; j < 16; ++j) dst[j] = (_Float16)0.f;
            }
        }
    };

    const v8f vzero = {0.f,0.f,0.f,0.f,0.f,0.f,0.f,0.f};
    v8f acc[4];
    acc[0] = vzero; acc[1] = vzero; acc[2] = vzero; acc[3] = vzero;

    const int mh = wave >> 2;    // pixel half of tile
    const int nq = wave & 3;     // channel quarter
    const int khi = lane >> 4;
    const int ml  = lane & 15;

    // ---- pipeline prologue ----
    stage(0, 0);
    gv_async_wait0();
    __syncthreads();

    for (int ic = 0; ic < nchunks; ++ic) {
        const int buf = ic & 1;
        // issue next chunk's DMA into the alternate buffer (overlaps WMMAs)
        if (ic + 1 < nchunks) stage((ic + 1) * 64, buf ^ 1);

        // ---- 2 x (4 WMMA): ISA 16-bit frag layout, K-contiguous in LDS ----
        #pragma unroll
        for (int s = 0; s < 2; ++s) {
            v16h bfrag;
            {
                const _Float16* br = &Bs[buf][nq * 16 + ml][s * 32];
                #pragma unroll
                for (int j = 0; j < 8; ++j) {
                    bfrag[j]     = br[8 * khi + j];
                    bfrag[8 + j] = br[16 + 8 * khi + j];
                }
            }
            #pragma unroll
            for (int t = 0; t < 4; ++t) {
                v16h afrag;
                const _Float16* ar = &As[buf][mh * 64 + t * 16 + ml][s * 32];
                #pragma unroll
                for (int j = 0; j < 8; ++j) {
                    afrag[j]     = ar[8 * khi + j];
                    afrag[8 + j] = ar[16 + 8 * khi + j];
                }
                acc[t] = __builtin_amdgcn_wmma_f32_16x16x32_f16(
                    false, afrag, false, bfrag, (short)0, acc[t], false, false);
            }
        }

        gv_async_wait0();      // next chunk's DMA landed
        __syncthreads();       // all waves done reading buf / writing buf^1
    }

    // ---- epilogue: per-lane channel scalars, coalesced NHWC f16 stores ----
    const int ch = n0 + nq * 16 + ml;
    const bool chOK = (ch < Cout);
    float scale = 1.f, shift = 0.f;
    if (chOK) {
        if (flags & F_BN) {
            const float inv = bng[ch] * rsqrtf(bnv[ch] + 1e-5f);
            scale = inv; shift = bnb[ch] - bnm[ch] * inv;
        }
        if (flags & F_BIAS) shift += bias[ch];
    }
    #pragma unroll
    for (int t = 0; t < 4; ++t) {
        #pragma unroll
        for (int r = 0; r < 8; ++r) {
            const int pix = m0 + mh * 64 + t * 16 + r + 8 * khi;
            if (chOK && pix < Npix) {
                float o = acc[t][r] * scale + shift;
                if (flags & F_RES)  o += (float)res[(long)pix * Cout + ch];
                if (flags & F_LEAKY) o = o > 0.f ? o : 0.1f * o;
                Y[(long)pix * Cout + ch] = (_Float16)o;
            }
        }
    }
}

// ---------------------------------------------------------------------------
__global__ void gv_maxpool2(const _Float16* __restrict__ X, _Float16* __restrict__ Y,
                            int B, int C, int H, int W)
{
    const int Ho = H >> 1, Wo = W >> 1;
    const long total = (long)B * Ho * Wo * C;
    const long i = (long)blockIdx.x * blockDim.x + threadIdx.x;
    if (i >= total) return;
    const int c  = (int)(i % C); long r = i / C;
    const int wo = (int)(r % Wo); r /= Wo;
    const int ho = (int)(r % Ho);
    const int b  = (int)(r / Ho);
    const long base = ((long)(b * H + 2 * ho) * W + 2 * wo) * C + c;
    const float a0 = (float)X[base],         a1 = (float)X[base + C];
    const float a2 = (float)X[base + (long)W * C], a3 = (float)X[base + (long)W * C + C];
    Y[i] = (_Float16)fmaxf(fmaxf(a0, a1), fmaxf(a2, a3));
}

// ---------------------------------------------------------------------------
// Soft-argmax: one wave per (b, channel); input NHWC f16 [B,10,10,32].
__global__ void gv_softargmax(const _Float16* __restrict__ Hh, float* __restrict__ pts,
                              int nrows)
{
    const int gw   = (blockIdx.x * blockDim.x + threadIdx.x) >> 5;
    const int lane = threadIdx.x & 31;
    if (gw >= nrows) return;
    const int b = gw >> 5, ch = gw & 31;
    const _Float16* f = Hh + (long)b * 100 * 32 + ch;
    float mx = -1e30f;
    for (int p = lane; p < 100; p += 32) mx = fmaxf(mx, (float)f[p * 32]);
    for (int o = 16; o; o >>= 1) mx = fmaxf(mx, __shfl_xor(mx, o, 32));
    float s = 0.f, sy = 0.f, sx = 0.f;
    for (int p = lane; p < 100; p += 32) {
        const float e = expf((float)f[p * 32] - mx);
        const int y = p / 10, x = p - 10 * y;
        s  += e;
        sy += e * (-1.f + 2.f * y / 9.f);
        sx += e * (-1.f + 2.f * x / 9.f);
    }
    for (int o = 16; o; o >>= 1) {
        s  += __shfl_xor(s,  o, 32);
        sy += __shfl_xor(sy, o, 32);
        sx += __shfl_xor(sx, o, 32);
    }
    if (lane == 0) { pts[gw * 2 + 0] = sy / s; pts[gw * 2 + 1] = sx / s; }
}

// ---------------------------------------------------------------------------
// Polygon soft-rasterization: one workgroup per mask (b,head); 8 points.
__global__ void gv_points_to_mask(const float* __restrict__ pts,
                                  float* __restrict__ mask)
{
    __shared__ float P0[8], P1[8], E0[8], E1[8];
    const int n = blockIdx.x;            // mask id = b*4 + head
    if (threadIdx.x == 0) {
        const int b = n >> 2, h = n & 3;
        float p0[8], p1[8], ang[8];
        int idx[8];
        float c0 = 0.f, c1 = 0.f;
        for (int k = 0; k < 8; ++k) {
            p0[k] = pts[((long)b * 32 + h * 8 + k) * 2 + 0];
            p1[k] = pts[((long)b * 32 + h * 8 + k) * 2 + 1];
            c0 += p0[k]; c1 += p1[k];
        }
        c0 *= 0.125f; c1 *= 0.125f;
        for (int k = 0; k < 8; ++k) { ang[k] = atan2f(p1[k] - c1, p0[k] - c0); idx[k] = k; }
        for (int i = 1; i < 8; ++i) {           // stable insertion argsort
            const float av = ang[i]; const int iv = idx[i];
            int j = i - 1;
            while (j >= 0 && ang[j] > av) { ang[j+1] = ang[j]; idx[j+1] = idx[j]; --j; }
            ang[j+1] = av; idx[j+1] = iv;
        }
        float s0[8], s1[8];
        for (int k = 0; k < 8; ++k) { s0[k] = p0[idx[k]]; s1[k] = p1[idx[k]]; }
        float signed_a = 0.f;
        for (int k = 0; k < 8; ++k) {
            const int kn = (k + 1) & 7;
            signed_a += s0[k] * s1[kn] - s1[k] * s0[kn];
        }
        if (signed_a < 0.f) {                    // orientation flip
            float t0[8], t1[8];
            for (int k = 0; k < 8; ++k) { t0[k] = s0[7 - k]; t1[k] = s1[7 - k]; }
            for (int k = 0; k < 8; ++k) { s0[k] = t0[k]; s1[k] = t1[k]; }
        }
        for (int k = 0; k < 8; ++k) {
            const int kn = (k + 1) & 7;
            P0[k] = s0[k]; P1[k] = s1[k];
            E0[k] = s0[kn] - s0[k]; E1[k] = s1[kn] - s1[k];
        }
    }
    __syncthreads();
    for (int cell = threadIdx.x; cell < 1024; cell += blockDim.x) {
        const int y = cell >> 5, x = cell & 31;
        const float g0 = -1.f + 2.f * y / 31.f;
        const float g1 = -1.f + 2.f * x / 31.f;
        float z[8], mx = -1e30f;
        #pragma unroll
        for (int k = 0; k < 8; ++k) {
            const float cr = E0[k] * (g1 - P1[k]) - E1[k] * (g0 - P0[k]);
            z[k] = -cr / GV_TAU;
            mx = fmaxf(mx, z[k]);
        }
        float s = 0.f;
        #pragma unroll
        for (int k = 0; k < 8; ++k) s += expf(z[k] - mx);
        const float lse = mx + logf(s);
        const float q = -(GV_TAU * lse) / GV_TAU;         // as written in reference
        mask[(long)n * 1024 + cell] = 1.f / (1.f + expf(-q));
    }
}

// ---------------------------------------------------------------------------
__global__ void gv_proto(const float* __restrict__ smask, const int* __restrict__ sy,
                         float* __restrict__ proto)
{
    const int i = blockIdx.x * blockDim.x + threadIdx.x;
    if (i >= 5 * 4 * 1024) return;
    const int cell = i & 1023;
    const int h    = (i >> 10) & 3;
    const int c    = i >> 12;
    float sum = 0.f; int cnt = 0;
    for (int s = 0; s < 25; ++s) {
        if (sy[s] == c) { sum += smask[((long)(s * 4 + h)) * 1024 + cell]; ++cnt; }
    }
    proto[i] = sum / (float)cnt;
}

// ---------------------------------------------------------------------------
__global__ void gv_iou(const float* __restrict__ qmask, const float* __restrict__ proto,
                       float* __restrict__ logits)
{
    const int id = blockIdx.x;           // q*5 + c
    const int q = id / 5, c = id - 5 * q;
    const int lane = threadIdx.x;
    float acc = 0.f;
    for (int h = 0; h < 4; ++h) {
        const float* qm = qmask + (long)(q * 4 + h) * 1024;
        const float* pm = proto + (long)(c * 4 + h) * 1024;
        float inter = 0.f, uni = 0.f;
        for (int p = lane; p < 1024; p += 32) {
            const float a = qm[p], b = pm[p], ab = a * b;
            inter += ab; uni += a + b - ab;
        }
        for (int o = 16; o; o >>= 1) {
            inter += __shfl_xor(inter, o, 32);
            uni   += __shfl_xor(uni,   o, 32);
        }
        acc += inter / (uni + 1e-6f);
    }
    if (lane == 0) logits[id] = acc * 0.25f;
}

// ---------------------------------------------------------------------------
__global__ void gv_finalize(const float* __restrict__ logits, const int* __restrict__ qy,
                            float* __restrict__ out)
{
    __shared__ float sl[128], sa[128];
    const int t = threadIdx.x;
    float lossv = 0.f, accv = 0.f;
    if (t < 75) {
        const float* row = logits + t * 5;
        float mx = row[0];
        for (int c = 1; c < 5; ++c) mx = fmaxf(mx, row[c]);
        float s = 0.f;
        for (int c = 0; c < 5; ++c) s += expf(row[c] - mx);
        const float lse = mx + logf(s);
        const int y = qy[t];
        lossv = -(row[y] - lse);
        int am = 0; float bv = row[0];
        for (int c = 1; c < 5; ++c) if (row[c] > bv) { bv = row[c]; am = c; }
        accv = (am == y) ? 1.f : 0.f;
    }
    sl[t] = lossv; sa[t] = accv;
    __syncthreads();
    for (int s = 64; s; s >>= 1) {
        if (t < s) { sl[t] += sl[t + s]; sa[t] += sa[t + s]; }
        __syncthreads();
    }
    if (t == 0) { out[0] = sl[0] / 75.f; out[1] = sa[0] / 75.f; }
    for (int i = t; i < 375; i += 128) out[2 + i] = logits[i];
}

// ---------------------------------------------------------------------------
extern "C" void kernel_launch(void* const* d_in, const int* in_sizes, int n_in,
                              void* d_out, int out_size, void* d_ws, size_t ws_size,
                              hipStream_t stream)
{
    (void)in_sizes; (void)n_in; (void)out_size; (void)ws_size;
    auto f32 = [&](int i) { return (const float*)d_in[i]; };

    // ---- workspace carve ----
    size_t off = 0;
    auto allocB = [&](size_t bytes) -> void* {
        void* p = (char*)d_ws + off;
        off += (bytes + 255) & ~(size_t)255;
        return p;
    };
    auto allocH = [&](size_t n) { return (_Float16*)allocB(n * sizeof(_Float16)); };
    auto allocF = [&](size_t n) { return (float*)allocB(n * sizeof(float)); };

    const size_t BIG = (size_t)100 * 64 * 84 * 84;   // largest activation (elems)
    _Float16* X0  = allocH((size_t)100 * 84 * 84 * 3);
    _Float16* T1  = allocH(BIG);
    _Float16* T2  = allocH(BIG);
    _Float16* SC  = allocH(BIG);
    _Float16* T3  = allocH(BIG);
    _Float16* PA  = allocH((size_t)100 * 64 * 42 * 42);
    _Float16* PB  = allocH((size_t)100 * 64 * 42 * 42);
    float*    PTS = allocF((size_t)100 * 32 * 2);
    float*    MSK = allocF((size_t)400 * 1024);
    float*    PRO = allocF((size_t)5 * 4 * 1024);
    float*    LOG = allocF((size_t)75 * 5);

    // packed-weight scratch
    const int CH[4] = {64, 128, 256, 512};
    _Float16* wpk[18];
    {
        int wi = 0, ci = 3;
        for (int bi = 0; bi < 4; ++bi) {
            const int co = CH[bi];
            wpk[wi++] = allocH((size_t)co * ci * 9);   // w1
            wpk[wi++] = allocH((size_t)co * co * 9);   // w2
            wpk[wi++] = allocH((size_t)co * ci);       // wsc
            wpk[wi++] = allocH((size_t)co * co * 9);   // w3
            ci = co;
        }
        wpk[16] = allocH((size_t)256 * 512);           // kp w1
        wpk[17] = allocH((size_t)32 * 256);            // kp w2
    }

    auto pack = [&](const float* w, _Float16* o, int co, int ci, int ksz) {
        const long n = (long)co * ci * ksz * ksz;
        gv_pack_w<<<(int)((n + 255) / 256), 256, 0, stream>>>(w, o, co, ci, ksz);
    };
    auto conv = [&](const _Float16* Xp, const _Float16* Wp, _Float16* Yp,
                    const float* g, const float* b, const float* m, const float* v,
                    const _Float16* res, const float* bias,
                    int Ci, int H, int Wd, int Co, int ksz, int flags) {
        const int Npix = 100 * H * Wd;
        const int cl2 = (Ci & (Ci - 1)) ? -1 : __builtin_ctz(Ci);
        dim3 grid((Npix + 127) / 128, (Co + 63) / 64);
        gv_conv_wmma<<<grid, dim3(256), 0, stream>>>(
            Xp, Wp, Yp, g, b, m, v, res, bias, 100, Ci, H, Wd, Co, ksz, cl2, flags);
    };

    // input -> NHWC f16
    {
        const long n = (long)100 * 84 * 84 * 3;
        gv_to_nhwc<<<(int)((n + 255) / 256), 256, 0, stream>>>(f32(87), f32(88), X0);
    }

    const _Float16* cur = X0;
    int H = 84, Wd = 84, Ci = 3;
    _Float16* poolbuf[2] = {PA, PB};
    for (int bi = 0; bi < 4; ++bi) {
        const int Co = CH[bi];
        const int base = bi * 20;
        _Float16 *W1 = wpk[bi*4+0], *W2 = wpk[bi*4+1], *WS = wpk[bi*4+2], *W3 = wpk[bi*4+3];
        pack(f32(base + 0),  W1, Co, Ci, 3);
        pack(f32(base + 5),  W2, Co, Co, 3);
        pack(f32(base + 15), WS, Co, Ci, 1);
        pack(f32(base + 10), W3, Co, Co, 3);

        conv(cur, W1, T1, f32(base + 1),  f32(base + 2),  f32(base + 3),  f32(base + 4),
             nullptr, nullptr, Ci, H, Wd, Co, 3, F_LEAKY | F_BN);
        conv(T1,  W2, T2, f32(base + 6),  f32(base + 7),  f32(base + 8),  f32(base + 9),
             nullptr, nullptr, Co, H, Wd, Co, 3, F_LEAKY | F_BN);
        conv(cur, WS, SC, f32(base + 16), f32(base + 17), f32(base + 18), f32(base + 19),
             nullptr, nullptr, Ci, H, Wd, Co, 1, F_BN);
        conv(T2,  W3, T3, f32(base + 11), f32(base + 12), f32(base + 13), f32(base + 14),
             SC, nullptr, Co, H, Wd, Co, 3, F_LEAKY | F_BN | F_RES);
        if (bi < 3) {
            const int Ho = H / 2, Wo = Wd / 2;
            const long total = (long)100 * Ho * Wo * Co;
            gv_maxpool2<<<(int)((total + 255) / 256), 256, 0, stream>>>(T3, poolbuf[bi & 1],
                                                                        100, Co, H, Wd);
            cur = poolbuf[bi & 1];
            H = Ho; Wd = Wo;
        } else {
            cur = T3;
        }
        Ci = Co;
    }

    // keypoint head
    pack(f32(80), wpk[16], 256, 512, 1);
    pack(f32(85), wpk[17], 32, 256, 1);
    conv(cur, wpk[16], T1, f32(81), f32(82), f32(83), f32(84),
         nullptr, nullptr, 512, 10, 10, 256, 1, F_LEAKY | F_BN);
    conv(T1, wpk[17], T2, nullptr, nullptr, nullptr, nullptr,
         nullptr, f32(86), 256, 10, 10, 32, 1, F_BIAS);

    // soft-argmax -> points [100,4,8,2]
    const int nrows = 100 * 32;
    gv_softargmax<<<(nrows * 32 + 127) / 128, 128, 0, stream>>>(T2, PTS, nrows);

    // polygon soft-masks
    gv_points_to_mask<<<400, 256, 0, stream>>>(PTS, MSK);

    // prototypes from support masks
    gv_proto<<<(5 * 4 * 1024 + 255) / 256, 256, 0, stream>>>(MSK, (const int*)d_in[89], PRO);

    // soft-IoU logits over query masks
    gv_iou<<<75 * 5, 32, 0, stream>>>(MSK + (size_t)100 * 1024, PRO, LOG);

    // loss / acc / logits -> d_out
    gv_finalize<<<1, 128, 0, stream>>>(LOG, (const int*)d_in[90], (float*)d_out);
}
